// ASDClassifier_10831907520910
// MI455X (gfx1250) — compile-verified, matrix operand order checked by reading, not run on hardware
//
#include <hip/hip_runtime.h>

// ---------------------------------------------------------------------------
// MI455X / gfx1250 implementation of the ASD classifier forward pass.
// wave32; all GEMMs via V_WMMA_F32_16X16X32_F16 (f16 in, f32 accumulate).
// Branch-free fragment loads (clamped addresses + cndmask), float4 vector
// loads on contiguous paths, compile-time transpose specialization.
// ---------------------------------------------------------------------------

typedef __attribute__((ext_vector_type(16))) _Float16 v16h;
typedef __attribute__((ext_vector_type(8)))  float    v8f;

#define BNS 0.9999950000374997f   // 1/sqrt(1+1e-5)
#define NB  16
#define NT  80
#define NV  200
#define POS 256000                // NB*NT*NV

__device__ __forceinline__ int kmap(int k0, int half, int i) {
  // K index of element i (0..15) of a v16h fragment:
  // lanes 0-15 hold K {0..7,16..23}, lanes 16-31 hold K {8..15,24..31}.
  return k0 + ((i < 8) ? (half * 8 + i) : (16 + half * 8 + (i - 8)));
}

__device__ __forceinline__ float actf(float x, int act) {
  switch (act) {
    case 1: return x > 0.f ? x : 0.f;                 // relu
    case 2: return x > 0.f ? x : 0.1f * x;            // leaky 0.1
    case 3: return x > 0.f ? x : (__expf(x) - 1.f);   // elu
  }
  return x;
}

// pack 16 floats (two 8-float runs) into a f16 fragment, zeroing when !ok
__device__ __forceinline__ void pack16(v16h& f, const float4& a0, const float4& a1,
                                       const float4& b0, const float4& b1, bool ok)
{
  f[0]=(_Float16)(ok?a0.x:0.f); f[1]=(_Float16)(ok?a0.y:0.f);
  f[2]=(_Float16)(ok?a0.z:0.f); f[3]=(_Float16)(ok?a0.w:0.f);
  f[4]=(_Float16)(ok?a1.x:0.f); f[5]=(_Float16)(ok?a1.y:0.f);
  f[6]=(_Float16)(ok?a1.z:0.f); f[7]=(_Float16)(ok?a1.w:0.f);
  f[8]=(_Float16)(ok?b0.x:0.f); f[9]=(_Float16)(ok?b0.y:0.f);
  f[10]=(_Float16)(ok?b0.z:0.f); f[11]=(_Float16)(ok?b0.w:0.f);
  f[12]=(_Float16)(ok?b1.x:0.f); f[13]=(_Float16)(ok?b1.y:0.f);
  f[14]=(_Float16)(ok?b1.z:0.f); f[15]=(_Float16)(ok?b1.w:0.f);
}

// relu(BN*x) variant for the folded conv input
__device__ __forceinline__ void pack16_relu(v16h& f, const float4& a0, const float4& a1,
                                            const float4& b0, const float4& b1, bool ok)
{
  f[0]=(_Float16)(ok?fmaxf(a0.x,0.f)*BNS:0.f); f[1]=(_Float16)(ok?fmaxf(a0.y,0.f)*BNS:0.f);
  f[2]=(_Float16)(ok?fmaxf(a0.z,0.f)*BNS:0.f); f[3]=(_Float16)(ok?fmaxf(a0.w,0.f)*BNS:0.f);
  f[4]=(_Float16)(ok?fmaxf(a1.x,0.f)*BNS:0.f); f[5]=(_Float16)(ok?fmaxf(a1.y,0.f)*BNS:0.f);
  f[6]=(_Float16)(ok?fmaxf(a1.z,0.f)*BNS:0.f); f[7]=(_Float16)(ok?fmaxf(a1.w,0.f)*BNS:0.f);
  f[8]=(_Float16)(ok?fmaxf(b0.x,0.f)*BNS:0.f); f[9]=(_Float16)(ok?fmaxf(b0.y,0.f)*BNS:0.f);
  f[10]=(_Float16)(ok?fmaxf(b0.z,0.f)*BNS:0.f); f[11]=(_Float16)(ok?fmaxf(b0.w,0.f)*BNS:0.f);
  f[12]=(_Float16)(ok?fmaxf(b1.x,0.f)*BNS:0.f); f[13]=(_Float16)(ok?fmaxf(b1.y,0.f)*BNS:0.f);
  f[14]=(_Float16)(ok?fmaxf(b1.z,0.f)*BNS:0.f); f[15]=(_Float16)(ok?fmaxf(b1.w,0.f)*BNS:0.f);
}

// contiguous fragment: row[k0 + lanepattern], row pre-clamped; K-tail wave-uniform
__device__ __forceinline__ v16h ld_contig(const float* __restrict__ row,
                                          int k0, int half, int K, bool ok)
{
  v16h f;
  if (k0 + 32 <= K) {     // wave-uniform
    const float4* p0 = (const float4*)(row + k0 + half * 8);
    const float4* p1 = (const float4*)(row + k0 + 16 + half * 8);
    pack16(f, p0[0], p0[1], p1[0], p1[1], ok);
  } else {
#pragma unroll
    for (int i = 0; i < 16; ++i) {
      int kk = kmap(k0, half, i);
      float v = row[kk < K ? kk : 0];
      f[i] = (_Float16)((ok && kk < K) ? v : 0.f);
    }
  }
  return f;
}

// strided fragment: P[kk*ld + fixed], fixed pre-clamped
__device__ __forceinline__ v16h ld_strided(const float* __restrict__ P, int ld, int fixed,
                                           int k0, int half, int K, bool ok)
{
  v16h f;
#pragma unroll
  for (int i = 0; i < 16; ++i) {
    int kk = kmap(k0, half, i);
    float v = P[(long)(kk < K ? kk : 0) * ld + fixed];
    f[i] = (_Float16)((ok && kk < K) ? v : 0.f);
  }
  return f;
}

// ---------------------------------------------------------------------------
// Generic batched WMMA GEMM:  C = act( oscale*( alpha * A@B + bias ) )
// A elem (m,k) = TA ? A[k*lda+m] : A[m*lda+k]   (+ batch*sA)
// B elem (k,n) = TB ? B[n*ldb+k] : B[k*ldb+n]   (+ batch*sB)
// One wave per 16x16 output tile; K zero-padded to 32.
// ---------------------------------------------------------------------------
template <bool TA, bool TB>
__global__ void k_wgemm(const float* __restrict__ A, const float* __restrict__ B,
                        const float* __restrict__ bias, float* __restrict__ C,
                        int M, int N, int K, int lda, int ldb, int ldc,
                        long sA, long sB, long sC, int nbatch,
                        float alpha, float oscale, int act)
{
  int lane = threadIdx.x & 31;
  long wid = ((long)blockIdx.x * blockDim.x + threadIdx.x) >> 5;
  int mt = (M + 15) >> 4, nt = (N + 15) >> 4;
  long tiles = (long)mt * nt * nbatch;
  if (wid >= tiles) return;
  int bidx = (int)(wid / ((long)mt * nt));
  int t    = (int)(wid % ((long)mt * nt));
  int m0 = (t / nt) * 16, n0 = (t % nt) * 16;
  const float* Ab = A + (long)bidx * sA;
  const float* Bb = B + (long)bidx * sB;
  int half = lane >> 4, l15 = lane & 15;
  int ar = m0 + l15;  bool arok = ar < M;  int arc = arok ? ar : 0;
  int bc = n0 + l15;  bool bcok = bc < N;  int bcc = bcok ? bc : 0;
  const float* Arow = Ab + (TA ? 0 : (long)arc * lda);
  const float* Brow = Bb + (TB ? (long)bcc * ldb : 0);
  v8f acc = {};
  for (int k0 = 0; k0 < K; k0 += 32) {
    v16h af, bf;
    if constexpr (TA) af = ld_strided(Ab, lda, arc, k0, half, K, arok);
    else              af = ld_contig(Arow, k0, half, K, arok);
    if constexpr (TB) bf = ld_contig(Brow, k0, half, K, bcok);
    else              bf = ld_strided(Bb, ldb, bcc, k0, half, K, bcok);
    acc = __builtin_amdgcn_wmma_f32_16x16x32_f16(false, af, false, bf,
                                                 (short)0, acc, false, false);
  }
  float* Cb = C + (long)bidx * sC;
  if (bcok) {
    float bv = bias ? bias[bc] : 0.f;
    if (m0 + 16 <= M) {
#pragma unroll
      for (int r = 0; r < 8; ++r)
        Cb[(long)(m0 + half * 8 + r) * ldc + bc] = actf(oscale * (alpha * acc[r] + bv), act);
    } else {
#pragma unroll
      for (int r = 0; r < 8; ++r) {
        int cm = m0 + half * 8 + r;
        if (cm < M) Cb[(long)cm * ldc + bc] = actf(oscale * (alpha * acc[r] + bv), act);
      }
    }
  }
}

// ---------------------------------------------------------------------------
// Fold {conv3,conv5,conv7,branch biases,BN,1x1 fusion} into one 7-tap conv.
// Weff[oc, dt*C+ic] (dt-major K so each 32-K slice has uniform dt), beff[oc].
// ---------------------------------------------------------------------------
__global__ void k_fold(const float* __restrict__ w3, const float* __restrict__ b3,
                       const float* __restrict__ w5, const float* __restrict__ b5,
                       const float* __restrict__ w7, const float* __restrict__ b7,
                       const float* __restrict__ fw, const float* __restrict__ fb2,
                       float* __restrict__ Weff, float* __restrict__ beff, int C)
{
  int K = C * 7;
  long tot = (long)C * K;
  for (long idx = (long)blockIdx.x * blockDim.x + threadIdx.x; idx < tot;
       idx += (long)gridDim.x * blockDim.x) {
    int oc = (int)(idx / K), k = (int)(idx % K);
    int dt = k / C, ic = k % C;
    float s = 0.f;
    for (int mc = 0; mc < C; ++mc) {
      float w = fw[oc * 3 * C + 2 * C + mc] * w7[((long)mc * C + ic) * 7 + dt];
      if (dt >= 1 && dt <= 5) w += fw[oc * 3 * C + C + mc] * w5[((long)mc * C + ic) * 5 + (dt - 1)];
      if (dt >= 2 && dt <= 4) w += fw[oc * 3 * C + mc]     * w3[((long)mc * C + ic) * 3 + (dt - 2)];
      s += w;
    }
    Weff[idx] = BNS * s;
  }
  for (int oc = blockIdx.x * blockDim.x + threadIdx.x; oc < C;
       oc += gridDim.x * blockDim.x) {
    float s = fb2[oc];
    for (int mc = 0; mc < C; ++mc)
      s += BNS * (fw[oc * 3 * C + mc] * b3[mc] + fw[oc * 3 * C + C + mc] * b5[mc] +
                  fw[oc * 3 * C + 2 * C + mc] * b7[mc]);
    beff[oc] = s;
  }
}

// ---------------------------------------------------------------------------
// Folded temporal conv via WMMA. Activations pos-major: act[((b*T+t)*V+v)*C+c].
// K = dt*IC+ic; dt uniform per 32-K slice, so the time-tap guard is scalar and
// the input gather is contiguous float4 loads of 8 consecutive channels.
// ---------------------------------------------------------------------------
__global__ void k_conv_wmma(const float* __restrict__ in, const float* __restrict__ Weff,
                            const float* __restrict__ beff, float* __restrict__ out,
                            int T, int V, int IC, int OC)
{
  int lane = threadIdx.x & 31;
  long wid = ((long)blockIdx.x * blockDim.x + threadIdx.x) >> 5;
  int vt = (V + 15) >> 4, ot = OC >> 4;
  long tiles = (long)NB * T * vt * ot;
  if (wid >= tiles) return;
  int oc0 = (int)(wid % ot) * 16;  long r = wid / ot;
  int v0 = (int)(r % vt) * 16;     r /= vt;
  int t = (int)(r % T), b = (int)(r / T);
  int K = IC * 7;
  int half = lane >> 4, l15 = lane & 15;
  int vv = v0 + l15;  bool vok = vv < V;  int vvc = vok ? vv : 0;
  int oc = oc0 + l15;
  const float* wrow = Weff + (long)oc * K;
  v8f acc = {};
  for (int k0 = 0; k0 < K; k0 += 32) {
    int dt = k0 / IC;                 // uniform within this K-slice
    int ic0 = k0 % IC;
    int tt = t + dt - 3;
    bool tok = (tt >= 0) && (tt < T); // wave-uniform
    const float* arow = in + ((((long)b * T + (tok ? tt : 0)) * V + vvc) * IC + ic0);
    v16h af, bf;
    {
      const float4* p0 = (const float4*)(arow + half * 8);
      const float4* p1 = (const float4*)(arow + 16 + half * 8);
      pack16_relu(af, p0[0], p0[1], p1[0], p1[1], vok && tok);
    }
    {
      const float4* p0 = (const float4*)(wrow + k0 + half * 8);
      const float4* p1 = (const float4*)(wrow + k0 + 16 + half * 8);
      pack16(bf, p0[0], p0[1], p1[0], p1[1], true);
    }
    acc = __builtin_amdgcn_wmma_f32_16x16x32_f16(false, af, false, bf,
                                                 (short)0, acc, false, false);
  }
  float bv = beff[oc];
  long obase = ((long)b * T + t) * V;
  if (v0 + 16 <= V) {
#pragma unroll
    for (int rr = 0; rr < 8; ++rr)
      out[(obase + v0 + half * 8 + rr) * OC + oc] = acc[rr] + bv;
  } else {
#pragma unroll
    for (int rr = 0; rr < 8; ++rr) {
      int vm = v0 + half * 8 + rr;
      if (vm < V) out[(obase + vm) * OC + oc] = acc[rr] + bv;
    }
  }
}

// ---------------------------------------------------------------------------
// GAT block 1 (C=1 rank-1 collapse): out[bt,i,f] = elu(wsum[bt,i]*W[f]).
// ---------------------------------------------------------------------------
__global__ void k_gat1(const float* __restrict__ x, const float* __restrict__ adj,
                       const float* __restrict__ gW, const float* __restrict__ ga,
                       float* __restrict__ out)
{
  __shared__ float sh[256], se2[256], sW[32], s12[2];
  int bt = blockIdx.x, b = bt / NT, t = bt % NT;
  if (threadIdx.x == 0) {
    float a1 = 0.f, a2 = 0.f;
    for (int f = 0; f < 32; ++f) { a1 += gW[f] * ga[f]; a2 += gW[f] * ga[32 + f]; }
    s12[0] = a1; s12[1] = a2;
  }
  if (threadIdx.x < 32) sW[threadIdx.x] = gW[threadIdx.x];
  for (int j = threadIdx.x; j < NV; j += blockDim.x)
    sh[j] = x[((long)b * NT + t) * NV + j];
  __syncthreads();
  float s2 = s12[1];
  for (int j = threadIdx.x; j < NV; j += blockDim.x) se2[j] = sh[j] * s2;
  __syncthreads();
  float s1 = s12[0];
  for (int i = threadIdx.x; i < NV; i += blockDim.x) {
    float e1 = sh[i] * s1;
    float m = -1e30f;
    for (int j = 0; j < NV; ++j) {
      float e = e1 + se2[j]; e = e > 0.f ? e : 0.2f * e;
      m = fmaxf(m, adj[i * NV + j] > 0.f ? e : -9e15f);
    }
    float ssum = 0.f, wsum = 0.f;
    for (int j = 0; j < NV; ++j) {
      float e = e1 + se2[j]; e = e > 0.f ? e : 0.2f * e;
      float p = adj[i * NV + j] > 0.f ? __expf(e - m) : 0.f;
      ssum += p; wsum += p * sh[j];
    }
    float ws = wsum / ssum;
    for (int f = 0; f < 32; ++f) {
      float y = ws * sW[f];
      out[((long)bt * NV + i) * 32 + f] = y > 0.f ? y : (__expf(y) - 1.f);
    }
  }
}

// GAT2: e1 = h.(W a1), e2 = h.(W a2)
__global__ void k_gat2_e(const float* __restrict__ h, const float* __restrict__ g2W,
                         const float* __restrict__ g2a,
                         float* __restrict__ e1, float* __restrict__ e2)
{
  __shared__ float wa1[32], wa2[32];
  if (threadIdx.x < 32) {
    int c = threadIdx.x; float a1 = 0.f, a2 = 0.f;
    for (int f = 0; f < 64; ++f) { a1 += g2W[c * 64 + f] * g2a[f]; a2 += g2W[c * 64 + f] * g2a[64 + f]; }
    wa1[c] = a1; wa2[c] = a2;
  }
  __syncthreads();
  long n = (long)NB * NT * NV;
  for (long idx = (long)blockIdx.x * blockDim.x + threadIdx.x; idx < n;
       idx += (long)gridDim.x * blockDim.x) {
    const float* hp = h + idx * 32;
    float a1 = 0.f, a2 = 0.f;
    for (int c = 0; c < 32; ++c) { a1 += hp[c] * wa1[c]; a2 += hp[c] * wa2[c]; }
    e1[idx] = a1; e2[idx] = a2;
  }
}

// GAT2 softmax row stats (max, sum) per (bt, i)
__global__ void k_gat2_stats(const float* __restrict__ e1, const float* __restrict__ e2,
                             const float* __restrict__ adj,
                             float* __restrict__ rmax, float* __restrict__ rsum)
{
  int bt = blockIdx.x;
  __shared__ float se2[256];
  for (int j = threadIdx.x; j < NV; j += blockDim.x) se2[j] = e2[(long)bt * NV + j];
  __syncthreads();
  for (int i = threadIdx.x; i < NV; i += blockDim.x) {
    float a1 = e1[(long)bt * NV + i];
    float m = -1e30f;
    for (int j = 0; j < NV; ++j) {
      float e = a1 + se2[j]; e = e > 0.f ? e : 0.2f * e;
      m = fmaxf(m, adj[i * NV + j] > 0.f ? e : -9e15f);
    }
    float s = 0.f;
    for (int j = 0; j < NV; ++j) {
      float e = a1 + se2[j]; e = e > 0.f ? e : 0.2f * e;
      s += adj[i * NV + j] > 0.f ? __expf(e - m) : 0.f;
    }
    rmax[(long)bt * NV + i] = m;
    rsum[(long)bt * NV + i] = s;
  }
}

// GAT2 (att @ h) via WMMA, attention probabilities built on the fly.
__global__ void k_gat2_attmm(const float* __restrict__ e1, const float* __restrict__ e2,
                             const float* __restrict__ adj, const float* __restrict__ rmax,
                             const float* __restrict__ rsum, const float* __restrict__ h,
                             float* __restrict__ Mout)
{
  int lane = threadIdx.x & 31;
  long wid = ((long)blockIdx.x * blockDim.x + threadIdx.x) >> 5;
  int vt = (NV + 15) >> 4;               // 13 i-tiles
  long tiles = (long)NB * NT * vt * 2;   // 32 channels -> 2 n-tiles
  if (wid >= tiles) return;
  int n0 = (int)(wid % 2) * 16;  long r = wid / 2;
  int i0 = (int)(r % vt) * 16;   int bt = (int)(r / vt);
  const float* e2b = e2 + (long)bt * NV;
  int half = lane >> 4, l15 = lane & 15;
  int ii = i0 + l15;  bool iok = ii < NV;  int iic = iok ? ii : 0;
  float a1  = e1[(long)bt * NV + iic];
  float mm  = rmax[(long)bt * NV + iic];
  float inv = 1.f / rsum[(long)bt * NV + iic];
  const float* adjrow = adj + (long)iic * NV;
  const float* hb = h + (long)bt * NV * 32;
  int c = n0 + l15;
  v8f acc = {};
  for (int k0 = 0; k0 < 224; k0 += 32) {
    v16h af, bf;
#pragma unroll
    for (int i = 0; i < 16; ++i) {
      int j = kmap(k0, half, i);
      int jc = j < NV ? j : 0;
      float av = adjrow[jc];
      float e = a1 + e2b[jc]; e = e > 0.f ? e : 0.2f * e;
      float p = __expf(e - mm) * inv;
      bool on = iok && (j < NV) && (av > 0.f);
      af[i] = (_Float16)(on ? p : 0.f);
    }
#pragma unroll
    for (int i = 0; i < 16; ++i) {
      int j = kmap(k0, half, i);
      int jc = j < NV ? j : 0;               // A is zero there; value irrelevant
      bf[i] = (_Float16)hb[(long)jc * 32 + c];
    }
    acc = __builtin_amdgcn_wmma_f32_16x16x32_f16(false, af, false, bf,
                                                 (short)0, acc, false, false);
  }
#pragma unroll
  for (int rr = 0; rr < 8; ++rr) {
    int im = i0 + half * 8 + rr;
    if (im < NV) Mout[((long)bt * NV + im) * 32 + c] = acc[rr];
  }
}

// 4-head attention over S=80, hd=16 (online softmax per query row)
__global__ void k_mha(const float* __restrict__ qkv, float* __restrict__ o)
{
  int b = blockIdx.x / 4, hh = blockIdx.x % 4;
  for (int s = threadIdx.x; s < NT; s += blockDim.x) {
    const float* q = qkv + ((long)(b * NT + s)) * 192 + hh * 16;
    float m = -1e30f, l = 0.f, av[16];
#pragma unroll
    for (int f = 0; f < 16; ++f) av[f] = 0.f;
    for (int t2 = 0; t2 < NT; ++t2) {
      const float* k = qkv + ((long)(b * NT + t2)) * 192 + 64 + hh * 16;
      const float* v = qkv + ((long)(b * NT + t2)) * 192 + 128 + hh * 16;
      float d = 0.f;
#pragma unroll
      for (int f = 0; f < 16; ++f) d += q[f] * k[f];
      d *= 0.25f;
      float mn = fmaxf(m, d);
      float corr = __expf(m - mn), p = __expf(d - mn);
      l = l * corr + p;
#pragma unroll
      for (int f = 0; f < 16; ++f) av[f] = av[f] * corr + p * v[f];
      m = mn;
    }
    float* op = o + ((long)(b * NT + s)) * 64 + hh * 16;
#pragma unroll
    for (int f = 0; f < 16; ++f) op[f] = av[f] / l;
  }
}

// y = layernorm(x + r) * g + b     (one thread per row, D=64)
__global__ void k_add_ln(const float* __restrict__ x, const float* __restrict__ r,
                         const float* __restrict__ g, const float* __restrict__ bb,
                         float* __restrict__ y, int rows)
{
  int i = blockIdx.x * blockDim.x + threadIdx.x;
  if (i >= rows) return;
  const float* xp = x + (long)i * 64; const float* rp = r + (long)i * 64;
  float mu = 0.f;
  for (int c = 0; c < 64; ++c) mu += xp[c] + rp[c];
  mu *= (1.f / 64.f);
  float var = 0.f;
  for (int c = 0; c < 64; ++c) { float d = xp[c] + rp[c] - mu; var += d * d; }
  var *= (1.f / 64.f);
  float is = rsqrtf(var + 1e-5f);
  float* yp = y + (long)i * 64;
  for (int c = 0; c < 64; ++c) yp[c] = (xp[c] + rp[c] - mu) * is * g[c] + bb[c];
}

__global__ void k_mean_t(const float* __restrict__ xs, float* __restrict__ feats, int ldf)
{
  int idx = blockIdx.x * blockDim.x + threadIdx.x;
  if (idx >= NB * 64) return;
  int b = idx / 64, c = idx % 64;
  float s = 0.f;
  for (int t = 0; t < NT; ++t) s += xs[((long)b * NT + t) * 64 + c];
  feats[(long)b * ldf + c] = s * (1.f / NT);
}

__global__ void k_center(const float* __restrict__ x, float* __restrict__ xc)
{
  int idx = blockIdx.x * blockDim.x + threadIdx.x;
  if (idx >= NB * NV) return;
  int b = idx / NV, n = idx % NV;
  float s = 0.f;
  for (int t = 0; t < NT; ++t) s += x[((long)b * NT + t) * NV + n];
  s *= (1.f / NT);
  for (int t = 0; t < NT; ++t)
    xc[((long)b * NT + t) * NV + n] = x[((long)b * NT + t) * NV + n] - s;
}

__global__ void k_init_eig(float* __restrict__ cov, float* __restrict__ U)
{
  long idx = (long)blockIdx.x * blockDim.x + threadIdx.x;
  long tot = (long)NB * NV * NV;
  if (idx >= tot) return;
  int rem = (int)(idx % (NV * NV)), i = rem / NV, j = rem % NV;
  if (i == j) cov[idx] += 1e-4f;
  U[idx] = (i == j) ? 1.f : 0.f;
}

// Cyclic Jacobi eigensolver; one block per 200x200 SPD matrix (L2-resident).
__global__ void k_jacobi(float* __restrict__ A, float* __restrict__ U, int sweeps)
{
  int b = blockIdx.x;
  float* Ab = A + (long)b * NV * NV;
  float* Ub = U + (long)b * NV * NV;
  __shared__ float cs[2];
  for (int sw = 0; sw < sweeps; ++sw) {
    for (int p = 0; p < NV - 1; ++p) {
      for (int q = p + 1; q < NV; ++q) {
        if (threadIdx.x == 0) {
          float apq = Ab[p * NV + q];
          if (fabsf(apq) < 1e-12f) { cs[0] = 1.f; cs[1] = 0.f; }
          else {
            float th = (Ab[q * NV + q] - Ab[p * NV + p]) / (2.f * apq);
            float tt = (th >= 0.f ? 1.f : -1.f) / (fabsf(th) + sqrtf(1.f + th * th));
            float c = rsqrtf(1.f + tt * tt);
            cs[0] = c; cs[1] = tt * c;
          }
        }
        __syncthreads();
        float c = cs[0], s = cs[1];
        if (s != 0.f) {
          for (int k = threadIdx.x; k < NV; k += blockDim.x) {
            float akp = Ab[k * NV + p], akq = Ab[k * NV + q];
            Ab[k * NV + p] = c * akp - s * akq;
            Ab[k * NV + q] = s * akp + c * akq;
            float ukp = Ub[k * NV + p], ukq = Ub[k * NV + q];
            Ub[k * NV + p] = c * ukp - s * ukq;
            Ub[k * NV + q] = s * ukp + c * ukq;
          }
          __syncthreads();
          for (int k = threadIdx.x; k < NV; k += blockDim.x) {
            float apk = Ab[p * NV + k], aqk = Ab[q * NV + k];
            Ab[p * NV + k] = c * apk - s * aqk;
            Ab[q * NV + k] = s * apk + c * aqk;
          }
        }
        __syncthreads();
      }
    }
  }
}

// Us[n,k] = U[n,k] * log(clip(lambda_k, 1e-6))
__global__ void k_scale_u(const float* __restrict__ A, const float* __restrict__ U,
                          float* __restrict__ Us)
{
  long idx = (long)blockIdx.x * blockDim.x + threadIdx.x;
  long tot = (long)NB * NV * NV;
  if (idx >= tot) return;
  int b = (int)(idx / (NV * NV)), rem = (int)(idx % (NV * NV)), k = rem % NV;
  float L = __logf(fmaxf(A[(long)b * NV * NV + k * NV + k], 1e-6f));
  Us[idx] = U[idx] * L;
}

__global__ void k_triu(const float* __restrict__ lc, float* __restrict__ tv)
{
  int i = blockIdx.x, b = blockIdx.y;
  long base = (long)i * NV - (long)i * (i - 1) / 2;
  for (int j = i + threadIdx.x; j < NV; j += blockDim.x)
    tv[(long)b * (NV * (NV + 1) / 2) + base + (j - i)] = lc[((long)b * NV + i) * NV + j];
}

// ---------------------------------------------------------------------------
static inline int wblocks(long waves) { return (int)((waves * 32 + 255) / 256); }

extern "C" void kernel_launch(void* const* d_in, const int* in_sizes, int n_in,
                              void* d_out, int out_size, void* d_ws, size_t ws_size,
                              hipStream_t stream) {
  (void)in_sizes; (void)n_in; (void)out_size; (void)ws_size;
  const float* x        = (const float*)d_in[0];
  const float* adj      = (const float*)d_in[1];
  const float* g1_W     = (const float*)d_in[2];
  const float* g1_a     = (const float*)d_in[3];
  const float* b1c3w    = (const float*)d_in[4];
  const float* b1c3b    = (const float*)d_in[5];
  const float* b1c5w    = (const float*)d_in[6];
  const float* b1c5b    = (const float*)d_in[7];
  const float* b1c7w    = (const float*)d_in[8];
  const float* b1c7b    = (const float*)d_in[9];
  const float* b1fw     = (const float*)d_in[10];
  const float* b1fb     = (const float*)d_in[11];
  const float* g2_W     = (const float*)d_in[12];
  const float* g2_a     = (const float*)d_in[13];
  const float* b2c3w    = (const float*)d_in[14];
  const float* b2c3b    = (const float*)d_in[15];
  const float* b2c5w    = (const float*)d_in[16];
  const float* b2c5b    = (const float*)d_in[17];
  const float* b2c7w    = (const float*)d_in[18];
  const float* b2c7b    = (const float*)d_in[19];
  const float* b2fw     = (const float*)d_in[20];
  const float* b2fb     = (const float*)d_in[21];
  const float* fb_w     = (const float*)d_in[22];
  const float* fb_b     = (const float*)d_in[23];
  const float* sp_w     = (const float*)d_in[24];
  const float* sp_b     = (const float*)d_in[25];
  const float* ain_w    = (const float*)d_in[26];
  const float* ain_b    = (const float*)d_in[27];
  const float* aout_w   = (const float*)d_in[28];
  const float* aout_b   = (const float*)d_in[29];
  const float* ln1_g    = (const float*)d_in[30];
  const float* ln1_b    = (const float*)d_in[31];
  const float* ln2_g    = (const float*)d_in[32];
  const float* ln2_b    = (const float*)d_in[33];
  const float* ff1_w    = (const float*)d_in[34];
  const float* ff1_b    = (const float*)d_in[35];
  const float* ff2_w    = (const float*)d_in[36];
  const float* ff2_b    = (const float*)d_in[37];
  const float* tn1_w    = (const float*)d_in[38];
  const float* tn1_b    = (const float*)d_in[39];
  const float* tn2_w    = (const float*)d_in[40];
  const float* tn2_b    = (const float*)d_in[41];
  const float* cl1_w    = (const float*)d_in[42];
  const float* cl1_b    = (const float*)d_in[43];
  const float* cl2_w    = (const float*)d_in[44];
  const float* cl2_b    = (const float*)d_in[45];
  const float* sc1_w    = (const float*)d_in[46];
  const float* sc1_b    = (const float*)d_in[47];
  const float* sc2_w    = (const float*)d_in[48];
  const float* sc2_b    = (const float*)d_in[49];
  float* out = (float*)d_out;
  float* ws  = (float*)d_ws;

  // Workspace layout (float offsets) with lifetime-based aliasing (~164 MB):
  float* h1   = ws + 0L;            // POS*32  (dead after conv1)
  float* c1   = ws + 8192000L;      // POS*32  (dead after attmm)
  float* Mb   = ws + 16384000L;     // POS*32
  float* h2   = ws + 24576000L;     // POS*64  (dead after conv2)
  float* c2   = ws + 0L;            // POS*64  (aliases h1/c1 region)
  float* e1b  = ws + 0L;            // aliases h1 (dead)
  float* e2b  = ws + 256000L;
  float* rmx  = ws + 512000L;
  float* rsm  = ws + 768000L;
  float* xs3  = ws + 24576000L;     // 1280*3200 (aliases h2, dead)
  float* S    = ws + 28672000L;     // 1280*64
  float* QKV  = ws + 28753920L;     // 1280*192
  float* O    = ws + 28999680L;     // 1280*64
  float* A1   = ws + 29081600L;
  float* xs4  = ws + 29163520L;
  float* F1   = ws + 29245440L;     // 1280*128
  float* F2   = ws + 29409280L;
  float* xs5  = ws + 29491200L;
  float* xc   = ws + 8192000L;      // tangent branch aliases c1/c2 (dead)
  float* cov  = ws + 8448000L;
  float* Umat = ws + 9088000L;
  float* Us   = ws + 9728000L;
  float* lcv  = ws + 10368000L;
  float* tv   = ws + 11008000L;     // 16*20100
  float* gbuf = ws + 11329600L;     // 16*256
  float* feats= ws + 11333696L;     // 16*128
  float* t1   = ws + 11335744L;     // 16*64
  float* t2   = ws + 11336768L;     // 16*32
  float* We1  = ws + 40960000L;     // 32*224
  float* be1  = ws + 40967168L;
  float* We2  = ws + 40967200L;     // 64*448
  float* be2  = ws + 40995872L;

  // 1-2. fold conv stacks
  k_fold<<<28, 256, 0, stream>>>(b1c3w, b1c3b, b1c5w, b1c5b, b1c7w, b1c7b, b1fw, b1fb, We1, be1, 32);
  k_fold<<<112, 256, 0, stream>>>(b2c3w, b2c3b, b2c5w, b2c5b, b2c7w, b2c7b, b2fw, b2fb, We2, be2, 64);
  // 3. GAT1 (rank-1)
  k_gat1<<<NB * NT, 256, 0, stream>>>(x, adj, g1_W, g1_a, h1);
  // 4. folded conv block 1 (WMMA)
  k_conv_wmma<<<wblocks((long)NB * NT * 13 * 2), 256, 0, stream>>>(h1, We1, be1, c1, NT, NV, 32, 32);
  // 5-7. GAT2: e-vectors, softmax stats, (att@h) WMMA
  k_gat2_e<<<512, 256, 0, stream>>>(c1, g2_W, g2_a, e1b, e2b);
  k_gat2_stats<<<NB * NT, 256, 0, stream>>>(e1b, e2b, adj, rmx, rsm);
  k_gat2_attmm<<<wblocks((long)NB * NT * 13 * 2), 256, 0, stream>>>(e1b, e2b, adj, rmx, rsm, c1, Mb);
  // 8. h2 = elu(M @ g2_W)
  k_wgemm<false, false><<<wblocks(16000L * 4), 256, 0, stream>>>(
      Mb, g2_W, nullptr, h2, POS, 64, 32, 32, 64, 64, 0, 0, 0, 1, 1.f, 1.f, 3);
  // 9. folded conv block 2 (WMMA)
  k_conv_wmma<<<wblocks((long)NB * NT * 13 * 4), 256, 0, stream>>>(h2, We2, be2, c2, NT, NV, 64, 64);
  // 10. fb: lrelu0.1(c2 @ fb_w^T + b)
  k_wgemm<false, true><<<wblocks(16000L), 256, 0, stream>>>(
      c2, fb_w, fb_b, xs3, POS, 16, 64, 64, 64, 16, 0, 0, 0, 1, 1.f, 1.f, 2);
  // 11. sp: (1280x3200) @ sp_w^T
  k_wgemm<false, true><<<wblocks(80L * 4), 256, 0, stream>>>(
      xs3, sp_w, sp_b, S, 1280, 64, 3200, 3200, 3200, 64, 0, 0, 0, 1, 1.f, 1.f, 0);
  // 12-15. MHA + residual LN
  k_wgemm<false, true><<<wblocks(80L * 12), 256, 0, stream>>>(
      S, ain_w, ain_b, QKV, 1280, 192, 64, 64, 64, 192, 0, 0, 0, 1, 1.f, 1.f, 0);
  k_mha<<<64, 128, 0, stream>>>(QKV, O);
  k_wgemm<false, true><<<wblocks(80L * 4), 256, 0, stream>>>(
      O, aout_w, aout_b, A1, 1280, 64, 64, 64, 64, 64, 0, 0, 0, 1, 1.f, 1.f, 0);
  k_add_ln<<<5, 256, 0, stream>>>(S, A1, ln1_g, ln1_b, xs4, 1280);
  // 16-18. FFN + residual LN
  k_wgemm<false, true><<<wblocks(80L * 8), 256, 0, stream>>>(
      xs4, ff1_w, ff1_b, F1, 1280, 128, 64, 64, 64, 128, 0, 0, 0, 1, 1.f, 1.f, 1);
  k_wgemm<false, true><<<wblocks(80L * 4), 256, 0, stream>>>(
      F1, ff2_w, ff2_b, F2, 1280, 64, 128, 128, 128, 64, 0, 0, 0, 1, 1.f, 1.f, 0);
  k_add_ln<<<5, 256, 0, stream>>>(xs4, F2, ln2_g, ln2_b, xs5, 1280);
  // 19. feat_st = mean over T -> feats[:, 0:64]
  k_mean_t<<<4, 256, 0, stream>>>(xs5, feats, 128);
  // 20-26. tangent-space branch
  k_center<<<13, 256, 0, stream>>>(x, xc);
  k_wgemm<true, false><<<wblocks(13L * 13 * NB), 256, 0, stream>>>(
      xc, xc, nullptr, cov, NV, NV, NT, NV, NV, NV, 16000, 16000, 40000, NB,
      1.f / (NT - 1), 1.f, 0);
  k_init_eig<<<2500, 256, 0, stream>>>(cov, Umat);
  k_jacobi<<<NB, 256, 0, stream>>>(cov, Umat, 8);
  k_scale_u<<<2500, 256, 0, stream>>>(cov, Umat, Us);
  k_wgemm<false, true><<<wblocks(13L * 13 * NB), 256, 0, stream>>>(
      Us, Umat, nullptr, lcv, NV, NV, NV, NV, NV, NV, 40000, 40000, 40000, NB,
      1.f, 1.f, 0);
  k_triu<<<dim3(NV, NB), 64, 0, stream>>>(lcv, tv);
  // 27-28. tn1 (BN-scaled lrelu0.1), tn2 relu -> feats[:, 64:128]
  k_wgemm<false, true><<<2, 256, 0, stream>>>(
      tv, tn1_w, tn1_b, gbuf, 16, 256, 20100, 20100, 20100, 256, 0, 0, 0, 1, 1.f, BNS, 2);
  k_wgemm<false, true><<<1, 256, 0, stream>>>(
      gbuf, tn2_w, tn2_b, feats + 64, 16, 64, 256, 256, 256, 128, 0, 0, 0, 1, 1.f, 1.f, 1);
  // 29-32. classifier + site heads -> d_out = [cls(16x2), site(16x17)]
  k_wgemm<false, true><<<1, 256, 0, stream>>>(
      feats, cl1_w, cl1_b, t1, 16, 64, 128, 128, 128, 64, 0, 0, 0, 1, 1.f, 1.f, 1);
  k_wgemm<false, true><<<1, 256, 0, stream>>>(
      t1, cl2_w, cl2_b, out, 16, 2, 64, 64, 64, 2, 0, 0, 0, 1, 1.f, 1.f, 0);
  k_wgemm<false, true><<<1, 256, 0, stream>>>(
      feats, sc1_w, sc1_b, t2, 16, 32, 128, 128, 128, 32, 0, 0, 0, 1, 1.f, 1.f, 1);
  k_wgemm<false, true><<<1, 256, 0, stream>>>(
      t2, sc2_w, sc2_b, out + 32, 16, 17, 32, 32, 32, 17, 0, 0, 0, 1, 1.f, 1.f, 0);
}
